// N2JNet_72851235274805
// MI455X (gfx1250) — matrix operand order repaired and checked by reading, not compiled, not run on hardware
//
#include <hip/hip_runtime.h>
#include <hip/hip_bf16.h>

// ---------------------------------------------------------------------------
// N2JNet forward for MI455X (gfx1250, wave32, WMMA 16x16x32 f16).
//
//   k_prep      : f32 weights -> f16, pre-swizzled into WMMA B-fragment order
//                 ([frag][lane][16 contiguous halves]) so each lane loads its
//                 whole fragment with two ds_load_b128.
//   k_init      : scalar f32 node-init MLP -> x in ws.
//   per iter (20x): memset agg; k_node (WMMA NodeModel + pre-aggr + atomic
//                 segment sum); k_global (scalar post-aggr, residual u).
//   k_out_local / k_out_global : scalar f32 output heads -> d_out.
// ---------------------------------------------------------------------------

typedef __attribute__((ext_vector_type(16))) _Float16 v16h;
typedef __attribute__((ext_vector_type(8)))  _Float16 v8h;
typedef __attribute__((ext_vector_type(8)))  float    v8f;

#define NTOT 262144
#define BSZ  8192
#define NIT  20

// f16 weight arena layout per iteration (units: halves).
// Each matrix is stored as fragments of 512 halves:
//   frag = (n0/16)*(Kpad/32) + (k0/32);  within frag: lane*16 + e,
//   value = W[k0 + (lane&16?8:0) + e + (e>=8?8:0)][n0 + (lane&15)]  (zero-padded).
#define NW1_OFF 0       // Kpad=64: 4 frags
#define NW2_OFF 2048    // Kpad=32: 2 frags
#define NW3_OFF 3072
#define PW1_OFF 4096
#define PW2_OFF 6144
#define PW3_OFF 7168
#define W_PER_IT 8192

// workspace byte offsets
#define WS_X    0u
#define WS_U    20971520u               // N*20*4
#define WS_AGG  21626880u               // + B*20*4
#define WS_WF   22282240u               // + B*20*4

// ---------------------------------------------------------------------------
// WMMA fragment helpers (layouts per CDNA5 ISA 7.12.2, wave32)
// ---------------------------------------------------------------------------

// A: 16x32 f16 tile from LDS row-major act[16][64] starting at k0.
// lane's data = two contiguous 8-half runs: K=k0+kb+0..7 and K=k0+kb+16..23
// (kb = 8 for lanes 16-31) -> two 16-byte LDS loads.
__device__ __forceinline__ v16h load_a(const _Float16* act, int k0, int lane) {
  int row = lane & 15;
  int kb  = (lane & 16) ? 8 : 0;
  const _Float16* p = act + row * 64 + k0 + kb;
  v8h lo = *(const v8h*)(p);
  v8h hi = *(const v8h*)(p + 16);
  v16h a;
#pragma unroll
  for (int i = 0; i < 8; ++i) { a[i] = lo[i]; a[8 + i] = hi[i]; }
  return a;
}

// B: pre-swizzled fragment -> 32 contiguous, 32B-aligned bytes per lane.
__device__ __forceinline__ v16h load_b(const _Float16* wmat, int frag, int lane) {
  return *(const v16h*)(wmat + frag * 512 + lane * 16);
}

// C/D: VGPR j holds row j (lanes 0-15) / row j+8 (lanes 16-31), col = lane&15.
__device__ __forceinline__ void store_c(float* out, int n0, int lane, v8f c) {
  int col = n0 + (lane & 15);
  int rb  = (lane & 16) ? 8 : 0;
#pragma unroll
  for (int j = 0; j < 8; ++j) out[(rb + j) * 32 + col] = c[j];
}

// out[16][32] = act[16][KPAD] @ W[KPAD][32]   (f16 in, f32 accum)
template <int KPAD>
__device__ __forceinline__ void gemm_tile(const _Float16* act, const _Float16* wmat,
                                          float* out, int lane) {
  constexpr int KFR = KPAD / 32;
  v16h a0 = load_a(act, 0, lane);
  v16h a1;
  if constexpr (KFR == 2) a1 = load_a(act, 32, lane);
#pragma unroll
  for (int nt = 0; nt < 2; ++nt) {
    v8f c = {};
    c = __builtin_amdgcn_wmma_f32_16x16x32_f16(false, a0, false,
                                               load_b(wmat, nt * KFR, lane),
                                               (short)0, c, false, false);
    if constexpr (KFR == 2)
      c = __builtin_amdgcn_wmma_f32_16x16x32_f16(false, a1, false,
                                                 load_b(wmat, nt * KFR + 1, lane),
                                                 (short)0, c, false, false);
    store_c(out, nt * 16, lane, c);
  }
}

// row-wise: y = [relu]( LN(c+b) * g + bt ), f16 out, zero-padded to col 32
__device__ __forceinline__ void row_bias_ln_relu(const float* C, const float* b,
                                                 const float* g, const float* bt,
                                                 _Float16* A, int lane, bool relu) {
  if (lane < 16) {
    const float* r = C + lane * 32;
    float t[20], m = 0.f;
#pragma unroll
    for (int j = 0; j < 20; ++j) { t[j] = r[j] + b[j]; m += t[j]; }
    m *= 0.05f;
    float v = 0.f;
#pragma unroll
    for (int j = 0; j < 20; ++j) { float d = t[j] - m; v += d * d; }
    float rs = rsqrtf(v * 0.05f + 1e-5f);
    _Float16* o = A + lane * 64;
#pragma unroll
    for (int j = 0; j < 20; ++j) {
      float y = (t[j] - m) * rs * g[j] + bt[j];
      o[j] = (_Float16)(relu ? fmaxf(y, 0.f) : y);
    }
#pragma unroll
    for (int j = 20; j < 32; ++j) o[j] = (_Float16)0.f;
  }
}

// row-wise: y = relu(c+b), f16 out, zero-padded to col 32
__device__ __forceinline__ void row_bias_relu(const float* C, const float* b,
                                              _Float16* A, int lane) {
  if (lane < 16) {
    const float* r = C + lane * 32;
    _Float16* o = A + lane * 64;
#pragma unroll
    for (int j = 0; j < 20; ++j) o[j] = (_Float16)fmaxf(r[j] + b[j], 0.f);
#pragma unroll
    for (int j = 20; j < 32; ++j) o[j] = (_Float16)0.f;
  }
}

// ---------------------------------------------------------------------------
// Weight prep: f32 [K][20] -> f16 B-fragment-swizzled, all 20 iterations
// ---------------------------------------------------------------------------
__global__ void k_prep(const float* __restrict__ nW1, const float* __restrict__ nW2,
                       const float* __restrict__ nW3, const float* __restrict__ pW1,
                       const float* __restrict__ pW2, const float* __restrict__ pW3,
                       _Float16* __restrict__ WF) {
  int idx = blockIdx.x * 256 + threadIdx.x;
  if (idx >= NIT * W_PER_IT) return;
  int it = idx >> 13;
  int r  = idx & (W_PER_IT - 1);
  const float* src; int off, Kact, kfr;
  if (r < 2048)      { src = nW1 + it * 800; off = r;        Kact = 40; kfr = 2; }
  else if (r < 3072) { src = nW2 + it * 400; off = r - 2048; Kact = 20; kfr = 1; }
  else if (r < 4096) { src = nW3 + it * 400; off = r - 3072; Kact = 20; kfr = 1; }
  else if (r < 6144) { src = pW1 + it * 800; off = r - 4096; Kact = 40; kfr = 2; }
  else if (r < 7168) { src = pW2 + it * 400; off = r - 6144; Kact = 20; kfr = 1; }
  else               { src = pW3 + it * 400; off = r - 7168; Kact = 20; kfr = 1; }
  int frag = off >> 9;
  int lane = (off >> 4) & 31;
  int e    = off & 15;
  int k0   = (frag % kfr) * 32;
  int n0   = (frag / kfr) * 16;
  int k    = k0 + ((lane & 16) ? 8 : 0) + e + ((e >= 8) ? 8 : 0);
  int col  = n0 + (lane & 15);
  float v = (col < 20 && k < Kact) ? src[k * 20 + col] : 0.f;
  WF[idx] = (_Float16)v;
}

// ---------------------------------------------------------------------------
// Node init MLP: 8 -> 20 -> 20 -> 20 + LN    (scalar f32, one thread/node)
// ---------------------------------------------------------------------------
__global__ void k_init(const float* __restrict__ x0,
                       const float* iW1, const float* ib1,
                       const float* iW2, const float* ib2,
                       const float* iW3, const float* ib3,
                       const float* g0, const float* bt0,
                       float* __restrict__ X) {
  int n = blockIdx.x * 256 + threadIdx.x;
  if (n >= NTOT) return;
  float in[8];
#pragma unroll
  for (int k = 0; k < 8; ++k) in[k] = x0[(size_t)n * 8 + k];
  float h1[20], h2[20], o[20];
  for (int j = 0; j < 20; ++j) {
    float s = ib1[j];
#pragma unroll
    for (int k = 0; k < 8; ++k) s += in[k] * iW1[k * 20 + j];
    h1[j] = fmaxf(s, 0.f);
  }
  for (int j = 0; j < 20; ++j) {
    float s = ib2[j];
    for (int k = 0; k < 20; ++k) s += h1[k] * iW2[k * 20 + j];
    h2[j] = fmaxf(s, 0.f);
  }
  float m = 0.f;
  for (int j = 0; j < 20; ++j) {
    float s = ib3[j];
    for (int k = 0; k < 20; ++k) s += h2[k] * iW3[k * 20 + j];
    o[j] = s; m += s;
  }
  m *= 0.05f;
  float v = 0.f;
  for (int j = 0; j < 20; ++j) { float d = o[j] - m; v += d * d; }
  float rs = rsqrtf(v * 0.05f + 1e-5f);
  for (int j = 0; j < 20; ++j)
    X[(size_t)n * 20 + j] = (o[j] - m) * rs * g0[j] + bt0[j];
}

// ---------------------------------------------------------------------------
// Node phase: one wave32 per 16-node tile; WMMA for all 6 GEMMs
// ---------------------------------------------------------------------------
__global__ void __launch_bounds__(256)
k_node(float* __restrict__ X, const float* __restrict__ U, float* __restrict__ AGG,
       const int* __restrict__ batch, const _Float16* __restrict__ WF, int it,
       const float* nb1, const float* ng1, const float* nB1,
       const float* nb2, const float* ng2, const float* nB2,
       const float* nb3, const float* ng3, const float* nB3,
       const float* pb1, const float* pb2, const float* pb3,
       const float* pg, const float* pB) {
  __shared__ __align__(32) _Float16 sW[W_PER_IT];  // this iter's 6 matrices
  __shared__ __align__(16) _Float16 sA[8][1024];   // per-wave A tile [16][64]
  __shared__ float    sC[8][512];                  // per-wave C tile [16][32]
  __shared__ _Float16 sUg[8][320];                 // per-wave gathered u [16][20]
  __shared__ int      sSeg[8][16];

  const int wave = threadIdx.x >> 5;
  const int lane = threadIdx.x & 31;

  // block-cooperative weight stage (16 KB)
  {
    const uint4* src = (const uint4*)(WF + (size_t)it * W_PER_IT);
    uint4* dst = (uint4*)sW;
    for (int i = threadIdx.x; i < W_PER_IT / 8; i += 256) dst[i] = src[i];
  }
  __syncthreads();

  _Float16* A  = sA[wave];
  float*    C  = sC[wave];
  _Float16* Ug = sUg[wave];
  int*      sg = sSeg[wave];

  const int base = (blockIdx.x * 8 + wave) * 16;

  const float* nb1i = nb1 + it * 20; const float* ng1i = ng1 + it * 20; const float* nB1i = nB1 + it * 20;
  const float* nb2i = nb2 + it * 20; const float* ng2i = ng2 + it * 20; const float* nB2i = nB2 + it * 20;
  const float* nb3i = nb3 + it * 20; const float* ng3i = ng3 + it * 20; const float* nB3i = nB3 + it * 20;
  const float* pb1i = pb1 + it * 20; const float* pb2i = pb2 + it * 20; const float* pb3i = pb3 + it * 20;
  const float* pgi  = pg  + it * 20; const float* pBi  = pB  + it * 20;

  if (lane < 16) sg[lane] = batch[base + lane];
  for (int i = lane; i < 320; i += 32) {
    int row = i / 20, col = i - row * 20;
    Ug[i] = (_Float16)U[(size_t)sg[row] * 20 + col];
  }
  // concat [x | u[batch] | 0-pad] as f16 A tile
  for (int i = lane; i < 1024; i += 32) {
    int row = i >> 6, col = i & 63;
    float v = 0.f;
    if (col < 20)      v = X[(size_t)(base + row) * 20 + col];
    else if (col < 40) v = (float)Ug[row * 20 + (col - 20)];
    A[i] = (_Float16)v;
  }

  // ---- NodeModel: Lin LN ReLU / Lin LN ReLU / Lin LN, residual ----
  gemm_tile<64>(A, sW + NW1_OFF, C, lane);
  row_bias_ln_relu(C, nb1i, ng1i, nB1i, A, lane, true);
  gemm_tile<32>(A, sW + NW2_OFF, C, lane);
  row_bias_ln_relu(C, nb2i, ng2i, nB2i, A, lane, true);
  gemm_tile<32>(A, sW + NW3_OFF, C, lane);

  if (lane < 16) {                         // residual + rebuild concat
    const float* r = C + lane * 32;
    float t[20], m = 0.f;
#pragma unroll
    for (int j = 0; j < 20; ++j) { t[j] = r[j] + nb3i[j]; m += t[j]; }
    m *= 0.05f;
    float v = 0.f;
#pragma unroll
    for (int j = 0; j < 20; ++j) { float d = t[j] - m; v += d * d; }
    float rs = rsqrtf(v * 0.05f + 1e-5f);
    _Float16* o = A + lane * 64;
#pragma unroll
    for (int j = 0; j < 20; ++j) {
      float xn = X[(size_t)(base + lane) * 20 + j] + (t[j] - m) * rs * ng3i[j] + nB3i[j];
      X[(size_t)(base + lane) * 20 + j] = xn;
      o[j] = (_Float16)xn;
    }
#pragma unroll
    for (int j = 20; j < 40; ++j) o[j] = Ug[lane * 20 + (j - 20)];
    // cols 40..63 still zero from initial fill
  }

  // ---- GlobalModel pre-aggr: Lin ReLU / Lin ReLU / Lin LN, segment-sum ----
  gemm_tile<64>(A, sW + PW1_OFF, C, lane);
  row_bias_relu(C, pb1i, A, lane);
  gemm_tile<32>(A, sW + PW2_OFF, C, lane);
  row_bias_relu(C, pb2i, A, lane);
  gemm_tile<32>(A, sW + PW3_OFF, C, lane);

  if (lane < 16) {
    const float* r = C + lane * 32;
    float t[20], m = 0.f;
#pragma unroll
    for (int j = 0; j < 20; ++j) { t[j] = r[j] + pb3i[j]; m += t[j]; }
    m *= 0.05f;
    float v = 0.f;
#pragma unroll
    for (int j = 0; j < 20; ++j) { float d = t[j] - m; v += d * d; }
    float rs = rsqrtf(v * 0.05f + 1e-5f);
    float* dst = AGG + (size_t)sg[lane] * 20;
#pragma unroll
    for (int j = 0; j < 20; ++j)
      atomicAdd(&dst[j], (t[j] - m) * rs * pgi[j] + pBi[j]);
  }
}

// ---------------------------------------------------------------------------
// Global phase: post-aggr MLP over graphs, residual on u  (scalar f32)
// ---------------------------------------------------------------------------
__global__ void k_global(const float* __restrict__ AGG, float* __restrict__ U, int it,
                         const float* qW1, const float* qb1,
                         const float* qW2, const float* qb2,
                         const float* qW3, const float* qb3,
                         const float* qg, const float* qB) {
  int g = blockIdx.x * 256 + threadIdx.x;
  if (g >= BSZ) return;
  const float* W1 = qW1 + it * 800; const float* b1 = qb1 + it * 20;
  const float* W2 = qW2 + it * 400; const float* b2 = qb2 + it * 20;
  const float* W3 = qW3 + it * 400; const float* b3 = qb3 + it * 20;
  const float* gg = qg + it * 20;   const float* gB = qB + it * 20;
  float in[40];
  for (int j = 0; j < 20; ++j) { in[j] = AGG[(size_t)g * 20 + j]; in[20 + j] = U[(size_t)g * 20 + j]; }
  float h1[20], h2[20], o[20];
  for (int j = 0; j < 20; ++j) {
    float s = b1[j];
    for (int k = 0; k < 40; ++k) s += in[k] * W1[k * 20 + j];
    h1[j] = fmaxf(s, 0.f);
  }
  for (int j = 0; j < 20; ++j) {
    float s = b2[j];
    for (int k = 0; k < 20; ++k) s += h1[k] * W2[k * 20 + j];
    h2[j] = fmaxf(s, 0.f);
  }
  float m = 0.f;
  for (int j = 0; j < 20; ++j) {
    float s = b3[j];
    for (int k = 0; k < 20; ++k) s += h2[k] * W3[k * 20 + j];
    o[j] = s; m += s;
  }
  m *= 0.05f;
  float v = 0.f;
  for (int j = 0; j < 20; ++j) { float d = o[j] - m; v += d * d; }
  float rs = rsqrtf(v * 0.05f + 1e-5f);
  for (int j = 0; j < 20; ++j)
    U[(size_t)g * 20 + j] += (o[j] - m) * rs * gg[j] + gB[j];
}

// ---------------------------------------------------------------------------
// Output heads (scalar f32)
// ---------------------------------------------------------------------------
__global__ void k_out_local(const float* __restrict__ X,
                            const float* W1, const float* b1,
                            const float* W2, const float* b2,
                            const float* W3, const float* b3,
                            float* __restrict__ out) {
  int n = blockIdx.x * 256 + threadIdx.x;
  if (n >= NTOT) return;
  float in[20], h1[20], h2[20];
  for (int j = 0; j < 20; ++j) in[j] = X[(size_t)n * 20 + j];
  for (int j = 0; j < 20; ++j) {
    float s = b1[j];
    for (int k = 0; k < 20; ++k) s += in[k] * W1[k * 20 + j];
    h1[j] = fmaxf(s, 0.f);
  }
  for (int j = 0; j < 20; ++j) {
    float s = b2[j];
    for (int k = 0; k < 20; ++k) s += h1[k] * W2[k * 20 + j];
    h2[j] = fmaxf(s, 0.f);
  }
  for (int j = 0; j < 4; ++j) {
    float s = b3[j];
    for (int k = 0; k < 20; ++k) s += h2[k] * W3[k * 4 + j];
    out[(size_t)n * 4 + j] = s;
  }
}

__global__ void k_out_global(const float* __restrict__ U,
                             const float* W1, const float* b1,
                             const float* W2, const float* b2,
                             const float* W3, const float* b3,
                             float* __restrict__ out) {
  int g = blockIdx.x * 256 + threadIdx.x;
  if (g >= BSZ) return;
  float in[20], h1[20], h2[20];
  for (int j = 0; j < 20; ++j) in[j] = U[(size_t)g * 20 + j];
  for (int j = 0; j < 20; ++j) {
    float s = b1[j];
    for (int k = 0; k < 20; ++k) s += in[k] * W1[k * 20 + j];
    h1[j] = fmaxf(s, 0.f);
  }
  for (int j = 0; j < 20; ++j) {
    float s = b2[j];
    for (int k = 0; k < 20; ++k) s += h1[k] * W2[k * 20 + j];
    h2[j] = fmaxf(s, 0.f);
  }
  for (int j = 0; j < 4; ++j) {
    float s = b3[j];
    for (int k = 0; k < 20; ++k) s += h2[k] * W3[k * 4 + j];
    out[(size_t)g * 4 + j] = s;
  }
}

// ---------------------------------------------------------------------------
extern "C" void kernel_launch(void* const* d_in, const int* in_sizes, int n_in,
                              void* d_out, int out_size, void* d_ws, size_t ws_size,
                              hipStream_t stream) {
  const float* x0    = (const float*)d_in[0];
  const int*   batch = (const int*)d_in[1];
  // init_params
  const float* iW1 = (const float*)d_in[3];  const float* ib1 = (const float*)d_in[4];
  const float* iW2 = (const float*)d_in[5];  const float* ib2 = (const float*)d_in[6];
  const float* iW3 = (const float*)d_in[7];  const float* ib3 = (const float*)d_in[8];
  const float* g0  = (const float*)d_in[9];  const float* bt0 = (const float*)d_in[10];
  // meta_params (dict insertion order)
  const float* nW1 = (const float*)d_in[11]; const float* nb1 = (const float*)d_in[12];
  const float* ng1 = (const float*)d_in[13]; const float* nB1 = (const float*)d_in[14];
  const float* nW2 = (const float*)d_in[15]; const float* nb2 = (const float*)d_in[16];
  const float* ng2 = (const float*)d_in[17]; const float* nB2 = (const float*)d_in[18];
  const float* nW3 = (const float*)d_in[19]; const float* nb3 = (const float*)d_in[20];
  const float* ng3 = (const float*)d_in[21]; const float* nB3 = (const float*)d_in[22];
  const float* pW1 = (const float*)d_in[23]; const float* pb1 = (const float*)d_in[24];
  const float* pW2 = (const float*)d_in[25]; const float* pb2 = (const float*)d_in[26];
  const float* pW3 = (const float*)d_in[27]; const float* pb3 = (const float*)d_in[28];
  const float* pg  = (const float*)d_in[29]; const float* pB  = (const float*)d_in[30];
  const float* qW1 = (const float*)d_in[31]; const float* qb1 = (const float*)d_in[32];
  const float* qW2 = (const float*)d_in[33]; const float* qb2 = (const float*)d_in[34];
  const float* qW3 = (const float*)d_in[35]; const float* qb3 = (const float*)d_in[36];
  const float* qg  = (const float*)d_in[37]; const float* qB  = (const float*)d_in[38];
  // output heads
  const float* oW1 = (const float*)d_in[39]; const float* ob1 = (const float*)d_in[40];
  const float* oW2 = (const float*)d_in[41]; const float* ob2 = (const float*)d_in[42];
  const float* oW3 = (const float*)d_in[43]; const float* ob3 = (const float*)d_in[44];
  const float* gW1 = (const float*)d_in[45]; const float* gb1 = (const float*)d_in[46];
  const float* gW2 = (const float*)d_in[47]; const float* gb2 = (const float*)d_in[48];
  const float* gW3 = (const float*)d_in[49]; const float* gb3 = (const float*)d_in[50];

  char* ws = (char*)d_ws;
  float*    X   = (float*)(ws + WS_X);
  float*    U   = (float*)(ws + WS_U);
  float*    AGG = (float*)(ws + WS_AGG);
  _Float16* WF  = (_Float16*)(ws + WS_WF);
  float*    out = (float*)d_out;

  k_prep<<<(NIT * W_PER_IT + 255) / 256, 256, 0, stream>>>(nW1, nW2, nW3, pW1, pW2, pW3, WF);
  k_init<<<NTOT / 256, 256, 0, stream>>>(x0, iW1, ib1, iW2, ib2, iW3, ib3, g0, bt0, X);
  hipMemsetAsync(U, 0, (size_t)BSZ * 20 * sizeof(float), stream);

  for (int it = 0; it < NIT; ++it) {
    hipMemsetAsync(AGG, 0, (size_t)BSZ * 20 * sizeof(float), stream);
    k_node<<<NTOT / 128, 256, 0, stream>>>(X, U, AGG, batch, WF, it,
        nb1, ng1, nB1, nb2, ng2, nB2, nb3, ng3, nB3,
        pb1, pb2, pb3, pg, pB);
    k_global<<<BSZ / 256, 256, 0, stream>>>(AGG, U, it,
        qW1, qb1, qW2, qb2, qW3, qb3, qg, qB);
  }

  k_out_local<<<NTOT / 256, 256, 0, stream>>>(X, oW1, ob1, oW2, ob2, oW3, ob3, out);
  k_out_global<<<BSZ / 256, 256, 0, stream>>>(U, gW1, gb1, gW2, gb2, gW3, gb3,
                                              out + (size_t)NTOT * 4);
}
